// SparseAttention_11905649344759
// MI455X (gfx1250) — matrix operand order repaired and checked by reading, not compiled, not run on hardware
//
#include <hip/hip_runtime.h>
#include <math.h>

typedef __attribute__((ext_vector_type(16))) _Float16 v16h;
typedef __attribute__((ext_vector_type(8)))  float    v8f;

#define GA_H 4
#define GA_E 32
#define GA_D 32

__device__ __forceinline__ void atomicMaxF32(float* addr, float val) {
    // works with init = -inf: positives via signed int max, negatives via unsigned min
    if (val >= 0.0f) atomicMax((int*)addr, __float_as_int(val));
    else             atomicMin((unsigned int*)addr, __float_as_uint(val));
}

// ---------------- init: out=0, denom=0, nmax=-inf ----------------
__global__ void ga_init_kernel(float* __restrict__ out, float* __restrict__ nmax,
                               float* __restrict__ denom, long long nOut, long long nSeg) {
    long long i = (long long)blockIdx.x * blockDim.x + threadIdx.x;
    if (i < nOut) out[i] = 0.0f;
    if (i < nSeg) { nmax[i] = -__builtin_inff(); denom[i] = 0.0f; }
}

// ---------------- qk via WMMA: 16 edge-dots per v_wmma ----------------
// wave handles (tile, head); A = Q rows of the 16 edges (16x32 f16),
// B = K rows of the 16 edges (32x16 f16, column n = edge n). Diagonal of
// D = A*B gives the 16 per-edge dot products.
__global__ __launch_bounds__(256) void ga_qk_wmma_kernel(
    const float* __restrict__ Q, const float* __restrict__ K,
    const int* __restrict__ src, const int* __restrict__ dst,
    float* __restrict__ qk, int numTiles, float scale) {
    const int wave = (int)((blockIdx.x * blockDim.x + threadIdx.x) >> 5);
    const int lane = threadIdx.x & 31;
    const int tile = wave / GA_H;
    const int head = wave % GA_H;
    if (tile >= numTiles) return;   // wave-uniform: EXEC all-1s for waves reaching WMMA

    const int r  = lane & 15;       // A: row M=r ; B: column N=r
    const int hi = lane >> 4;
    const int e  = tile * 16 + r;
    const int es = src[e];
    const int ed = dst[e];
    const float* __restrict__ qrow = Q + ((size_t)es * GA_H + head) * GA_E;
    const float* __restrict__ krow = K + ((size_t)ed * GA_H + head) * GA_E;

    v16h a, b;
#pragma unroll
    for (int v = 0; v < 8; ++v) {
        // A 16x32 f16 layout (ISA 7.12.2): VGPR v<4: K=2v+8*hi ; v>=4: K=16+2(v-4)+8*hi
        const int ka = ((v >> 2) << 4) + (hi << 3) + ((v & 3) << 1);
        a[2 * v]     = (_Float16)qrow[ka];
        a[2 * v + 1] = (_Float16)qrow[ka + 1];
        // B 32x16 f16 layout: same K order as A: v<4: K=2v+8*hi ; v>=4: K=16+2(v-4)+8*hi
        const int kb = ((v >> 2) << 4) + (hi << 3) + ((v & 3) << 1);
        b[2 * v]     = (_Float16)krow[kb];
        b[2 * v + 1] = (_Float16)krow[kb + 1];
    }

    v8f c = {};
    c = __builtin_amdgcn_wmma_f32_16x16x32_f16(false, a, false, b, (short)0, c, false, false);

    // diagonal: i<8 -> (lane=i, vgpr=i); i>=8 -> (lane=i+8, vgpr=i-8)
    int sel = -1;
    if (lane < 8) sel = lane;
    else if (lane >= 16 && lane < 24) sel = lane - 16;
    float d = 0.0f;
#pragma unroll
    for (int i = 0; i < 8; ++i) d = (sel == i) ? c[i] : d;
    if (sel >= 0) {
        const int eo = tile * 16 + ((lane < 8) ? lane : (lane - 8));
        qk[(size_t)eo * GA_H + head] = d * scale;
    }
}

// scalar tail for NE % 16 edges (NE=640000 is divisible; kept for generality)
__global__ void ga_qk_tail_kernel(const float* __restrict__ Q, const float* __restrict__ K,
                                  const int* __restrict__ src, const int* __restrict__ dst,
                                  float* __restrict__ qk, long long eBase, long long nWork,
                                  float scale) {
    long long i = (long long)blockIdx.x * blockDim.x + threadIdx.x;
    if (i >= nWork) return;
    long long e = eBase + i / GA_H;
    int h = (int)(i % GA_H);
    const float* qrow = Q + ((size_t)src[e] * GA_H + h) * GA_E;
    const float* krow = K + ((size_t)dst[e] * GA_H + h) * GA_E;
    float acc = 0.0f;
#pragma unroll
    for (int k = 0; k < GA_E; ++k) acc += qrow[k] * krow[k];
    qk[(size_t)e * GA_H + h] = acc * scale;
}

// ---------------- segment max over edges grouped by src ----------------
__global__ void ga_segmax_kernel(const float* __restrict__ qk, const int* __restrict__ src,
                                 float* __restrict__ nmax, long long nEH) {
    long long i = (long long)blockIdx.x * blockDim.x + threadIdx.x;
    if (i >= nEH) return;
    int s = src[i / GA_H];
    atomicMaxF32(&nmax[(size_t)s * GA_H + (i % GA_H)], qk[i]);
}

// ---------------- ex = exp(qk - max[src]); denom += ex ----------------
__global__ void ga_exp_kernel(float* __restrict__ qk, const int* __restrict__ src,
                              const float* __restrict__ nmax, float* __restrict__ denom,
                              long long nEH) {
    long long i = (long long)blockIdx.x * blockDim.x + threadIdx.x;
    if (i >= nEH) return;
    int s = src[i / GA_H];
    size_t seg = (size_t)s * GA_H + (i % GA_H);
    float ex = __expf(qk[i] - nmax[seg]);
    qk[i] = ex;
    atomicAdd(&denom[seg], ex);
}

// ---------------- out[src] += (ex/denom[src]) * V[dst] ----------------
__global__ __launch_bounds__(256) void ga_agg_kernel(
    const float* __restrict__ qk, const float* __restrict__ denom,
    const float* __restrict__ V, const int* __restrict__ src, const int* __restrict__ dst,
    float* __restrict__ out, long long nEH) {
    long long wave = ((long long)blockIdx.x * blockDim.x + threadIdx.x) >> 5;
    int lane = threadIdx.x & 31;   // lane = d channel
    if (wave >= nEH) return;
    long long e = wave / GA_H;
    int h = (int)(wave % GA_H);
    int es = src[e], ed = dst[e];
    float alpha = qk[(size_t)e * GA_H + h] / denom[(size_t)es * GA_H + h];
    float v = V[((size_t)ed * GA_H + h) * GA_D + lane];
    atomicAdd(&out[((size_t)es * GA_H + h) * GA_D + lane], alpha * v);
}

extern "C" void kernel_launch(void* const* d_in, const int* in_sizes, int n_in,
                              void* d_out, int out_size, void* d_ws, size_t ws_size,
                              hipStream_t stream) {
    const float* Q   = (const float*)d_in[0];
    const float* K   = (const float*)d_in[1];
    const float* V   = (const float*)d_in[2];
    const int*   adj = (const int*)d_in[3];
    float* out = (float*)d_out;

    const long long L  = (long long)in_sizes[0] / (GA_H * GA_E);
    const long long NE = (long long)in_sizes[3] / 2;
    const int* src = adj;
    const int* dst = adj + NE;

    // workspace: qk/ex (NE*H) | nmax (L*H) | denom (L*H)
    float* qk    = (float*)d_ws;
    float* nmax  = qk + (size_t)NE * GA_H;
    float* denom = nmax + (size_t)L * GA_H;

    const float scale = 1.0f / sqrtf((float)GA_E);
    const long long nEH  = NE * GA_H;
    const long long nOut = L * GA_H * GA_D;
    const long long nSeg = L * GA_H;

    {   // init accumulators (every call: harness does not re-poison between replays)
        long long n = (nOut > nSeg) ? nOut : nSeg;
        ga_init_kernel<<<(int)((n + 255) / 256), 256, 0, stream>>>(out, nmax, denom, nOut, nSeg);
    }
    {   // qk via WMMA: 16 edges per matrix op
        long long numTiles = NE / 16;
        long long waves = numTiles * GA_H;
        ga_qk_wmma_kernel<<<(int)((waves + 7) / 8), 256, 0, stream>>>(
            Q, K, src, dst, qk, (int)numTiles, scale);
        long long tailEdges = NE - numTiles * 16;
        if (tailEdges > 0) {
            long long nWork = tailEdges * GA_H;
            ga_qk_tail_kernel<<<(int)((nWork + 255) / 256), 256, 0, stream>>>(
                Q, K, src, dst, qk, numTiles * 16, nWork, scale);
        }
    }
    ga_segmax_kernel<<<(int)((nEH + 255) / 256), 256, 0, stream>>>(qk, src, nmax, nEH);
    ga_exp_kernel<<<(int)((nEH + 255) / 256), 256, 0, stream>>>(qk, src, nmax, denom, nEH);
    ga_agg_kernel<<<(int)((nEH + 7) / 8), 256, 0, stream>>>(qk, denom, V, src, dst, out, nEH);
}
